// TaskTaskLayer_7095285973619
// MI455X (gfx1250) — matrix operand order repaired and checked by reading, not compiled, not run on hardware
//
#include <hip/hip_runtime.h>
#include <hip/hip_bf16.h>

#define NN 50000   // 3125 * 16
#define NE 800000  // 100000 * 8
#define DIM 128

typedef __attribute__((ext_vector_type(16))) _Float16 v16h;
typedef __attribute__((ext_vector_type(8)))  float    v8f;

// ---------- order-preserving float <-> uint map for atomic max ----------
__device__ __forceinline__ unsigned flipf(float f) {
    unsigned u = __float_as_uint(f);
    return (u & 0x80000000u) ? ~u : (u | 0x80000000u);
}
__device__ __forceinline__ float unflipf(unsigned u) {
    u = (u & 0x80000000u) ? (u & 0x7FFFFFFFu) : ~u;
    return __uint_as_float(u);
}

// ---------- prep: transpose+convert [Wsrc|Wres] -> Wt[256][128] f16 ----------
__global__ __launch_bounds__(256) void k_prep_wt(const float* __restrict__ Ws,
                                                 const float* __restrict__ Wr,
                                                 _Float16* __restrict__ Wt) {
    int idx  = blockIdx.x * 256 + threadIdx.x;     // 0..32767
    int part = idx >> 14;                          // 0: Wsrc, 1: Wres
    int r    = idx & 16383;
    int c    = r >> 7;                             // output col 0..127
    int k    = r & 127;                            // K index
    const float* W = part ? Wr : Ws;
    Wt[(part * 128 + c) * 128 + k] = (_Float16)W[k * 128 + c];
}

// ---------- prep: vdst[k] = sum_c Wdst[k][c]*att_dst[c];  kedge = We . att_e ----------
__global__ void k_prep_attn(const float* __restrict__ Wd, const float* __restrict__ ad,
                            const float* __restrict__ We, const float* __restrict__ ae,
                            float* __restrict__ vdst, float* __restrict__ kedge) {
    int k = threadIdx.x;                           // 128 threads
    float s = 0.f;
    for (int c = 0; c < 128; ++c) s += Wd[k * 128 + c] * ad[c];
    vdst[k] = s;
    if (k == 0) {
        float e = 0.f;
        for (int c = 0; c < 128; ++c) e += We[c] * ae[c];
        kedge[0] = e;
    }
}

// ---------- WMMA GEMM: H[N][256] = x[N][128] (f32->f16) @ Wt^T ----------
// N must be a multiple of 16 (NN = 3125*16); grid covers rows exactly.
__global__ __launch_bounds__(256) void k_gemm(const float* __restrict__ x,
                                              const _Float16* __restrict__ Wt,
                                              float* __restrict__ H) {
    __shared__ __align__(16) _Float16 As[16 * 128];
    const int row0 = blockIdx.x * 16;
    const int tid  = threadIdx.x;

    // load 16x128 f32 tile -> f16 LDS (512 float4 loads, 2 per thread)
    #pragma unroll
    for (int i = 0; i < 2; ++i) {
        int idx = tid + i * 256;
        int r   = idx >> 5;
        int c4  = idx & 31;
        float4 f = ((const float4*)x)[(row0 + r) * 32 + c4];
        _Float16* d = &As[r * 128 + c4 * 4];
        d[0] = (_Float16)f.x; d[1] = (_Float16)f.y;
        d[2] = (_Float16)f.z; d[3] = (_Float16)f.w;
    }
    __syncthreads();

    const int wave = tid >> 5;
    const int lane = tid & 31;
    const int m    = lane & 15;
    const int half = lane >> 4;

    // A fragments: identical for both col-tiles of this wave -> load once.
    // Layout (16x32 f16): VGPR0-3 hold K = kk + half*8 + 0..7,
    //                     VGPR4-7 hold K = kk + 16 + half*8 + 0..7
    union { v16h v; uint4 q[2]; } a[4];
    #pragma unroll
    for (int s = 0; s < 4; ++s) {
        const int kk = s * 32;
        a[s].q[0] = *(const uint4*)&As[m * 128 + kk + half * 8];
        a[s].q[1] = *(const uint4*)&As[m * 128 + kk + 16 + half * 8];
    }

    #pragma unroll
    for (int t = 0; t < 2; ++t) {
        const int ct = wave * 2 + t;               // col tile 0..15
        const int n  = ct * 16 + m;                // global col 0..255
        v8f c = {};
        #pragma unroll
        for (int s = 0; s < 4; ++s) {
            const int kk = s * 32;
            // B 32x16 layout: same as A, K = kk + half*8 + 0..7 and kk + 16 + half*8 + 0..7 in Wt[n][:]
            union { v16h v; uint4 q[2]; } b;
            const uint4* bp = (const uint4*)&Wt[n * 128 + kk + half * 8];
            b.q[0] = bp[0];
            b.q[1] = bp[2];
            c = __builtin_amdgcn_wmma_f32_16x16x32_f16(
                    false, a[s].v, false, b.v, (short)0, c, false, false);
        }
        // D layout: VGPR r -> row (r + 8*half), col = ct*16 + (lane&15)
        float* outp = &H[(row0 + 8 * half) * 256 + ct * 16 + m];
        #pragma unroll
        for (int r = 0; r < 8; ++r) outp[r * 256] = c[r];
    }
}

// ---------- per-node attention scalars: a_src = hs.att_src ; a_dst = x.vdst ----------
__global__ __launch_bounds__(256) void k_node_attn(const float* __restrict__ x,
                                                   const float* __restrict__ H,
                                                   const float* __restrict__ att_src,
                                                   const float* __restrict__ vdst,
                                                   float* __restrict__ a_src,
                                                   float* __restrict__ a_dst) {
    int node = blockIdx.x * 8 + (threadIdx.x >> 5);
    int lane = threadIdx.x & 31;
    float4 hs = ((const float4*)H)[node * 64 + lane];      // cols 0..127 of H
    float4 as = ((const float4*)att_src)[lane];
    float4 xv = ((const float4*)x)[node * 32 + lane];
    float4 vd = ((const float4*)vdst)[lane];
    float s = hs.x * as.x + hs.y * as.y + hs.z * as.z + hs.w * as.w;
    float d = xv.x * vd.x + xv.y * vd.y + xv.z * vd.z + xv.w * vd.w;
    for (int m = 16; m; m >>= 1) {
        s += __shfl_xor(s, m, 32);
        d += __shfl_xor(d, m, 32);
    }
    if (lane == 0) { a_src[node] = s; a_dst[node] = d; }
}

// ---------- zero/init scratch per conv ----------
__global__ void k_init(float* __restrict__ aggr, float* __restrict__ denom,
                       unsigned* __restrict__ maxbuf) {
    int i = blockIdx.x * 256 + threadIdx.x;        // grid covers NN*128 exactly
    aggr[i] = 0.f;
    if (i < NN) { denom[i] = 0.f; maxbuf[i] = flipf(-INFINITY); }
}

// ---------- edge pass 1: alpha + segment max ----------
__global__ void k_edge_alpha(const int* __restrict__ src, const int* __restrict__ dst,
                             const float* __restrict__ eattr,
                             const float* __restrict__ a_src, const float* __restrict__ a_dst,
                             const float* __restrict__ kedge,
                             float* __restrict__ alpha, unsigned* __restrict__ maxbuf) {
    int e = blockIdx.x * 256 + threadIdx.x;        // grid covers NE exactly
    int s = src[e], d = dst[e];
    float al = a_src[s] + a_dst[d] + kedge[0] * eattr[e];
    al = (al > 0.f) ? al : 0.2f * al;              // attention LeakyReLU
    alpha[e] = al;
    atomicMax(&maxbuf[d], flipf(al));
}

// ---------- edge pass 2: exp + segment sum ----------
__global__ void k_edge_exp(const int* __restrict__ dst, float* __restrict__ alpha,
                           const unsigned* __restrict__ maxbuf,
                           float* __restrict__ denom) {
    int e = blockIdx.x * 256 + threadIdx.x;
    int d = dst[e];
    float ea = __expf(alpha[e] - unflipf(maxbuf[d]));
    alpha[e] = ea;
    atomicAdd(&denom[d], ea);
}

// ---------- edge pass 3: weighted message scatter ----------
__global__ __launch_bounds__(256) void k_edge_scatter(const int* __restrict__ src,
                                                      const int* __restrict__ dst,
                                                      const float* __restrict__ ealpha,
                                                      const float* __restrict__ denom,
                                                      const float* __restrict__ H,
                                                      float* __restrict__ aggr) {
    int e    = blockIdx.x * 8 + (threadIdx.x >> 5);   // grid covers NE exactly
    int lane = threadIdx.x & 31;
    int s = src[e], d = dst[e];
    float den  = denom[d];
    float coef = ealpha[e] / ((den > 0.f) ? den : 1.f);
    float4 h = ((const float4*)H)[s * 64 + lane];     // hs = H cols 0..127
    float* base = &aggr[d * 128 + lane * 4];
    atomicAdd(base + 0, h.x * coef);
    atomicAdd(base + 1, h.y * coef);
    atomicAdd(base + 2, h.z * coef);
    atomicAdd(base + 3, h.w * coef);
}

// ---------- epilogue: aggr + hres + bias (+ LN + act) ----------
__global__ __launch_bounds__(256) void k_node_out(const float* __restrict__ aggr,
                                                  const float* __restrict__ H,
                                                  const float* __restrict__ bias,
                                                  const float* __restrict__ lng,
                                                  const float* __restrict__ lnb,
                                                  int do_ln, float* __restrict__ out,
                                                  int stride_f4, int col_f4) {
    int node = blockIdx.x * 8 + (threadIdx.x >> 5);   // grid covers NN exactly
    int lane = threadIdx.x & 31;
    float4 a = ((const float4*)aggr)[node * 32 + lane];
    float4 r = ((const float4*)H)[node * 64 + 32 + lane];  // hres = H cols 128..255
    float4 b = ((const float4*)bias)[lane];
    float4 o;
    o.x = a.x + r.x + b.x; o.y = a.y + r.y + b.y;
    o.z = a.z + r.z + b.z; o.w = a.w + r.w + b.w;
    if (do_ln) {
        float s  = o.x + o.y + o.z + o.w;
        float ss = o.x * o.x + o.y * o.y + o.z * o.z + o.w * o.w;
        for (int m = 16; m; m >>= 1) {
            s  += __shfl_xor(s, m, 32);
            ss += __shfl_xor(ss, m, 32);
        }
        float mu  = s * (1.f / 128.f);
        float var = ss * (1.f / 128.f) - mu * mu;
        float inv = rsqrtf(var + 1e-5f);
        float4 g  = ((const float4*)lng)[lane];
        float4 bb = ((const float4*)lnb)[lane];
        o.x = (o.x - mu) * inv * g.x + bb.x;
        o.y = (o.y - mu) * inv * g.y + bb.y;
        o.z = (o.z - mu) * inv * g.z + bb.z;
        o.w = (o.w - mu) * inv * g.w + bb.w;
        o.x = (o.x > 0.f) ? o.x : 0.01f * o.x;
        o.y = (o.y > 0.f) ? o.y : 0.01f * o.y;
        o.z = (o.z > 0.f) ? o.z : 0.01f * o.z;
        o.w = (o.w > 0.f) ? o.w : 0.01f * o.w;
    }
    ((float4*)out)[node * stride_f4 + col_f4 + lane] = o;
}

// =======================================================================
struct ConvP {
    const float *Wsrc, *Wdst, *Wedge, *asrc, *adst, *aedge, *Wres, *bias;
};

extern "C" void kernel_launch(void* const* d_in, const int* in_sizes, int n_in,
                              void* d_out, int out_size, void* d_ws, size_t ws_size,
                              hipStream_t stream) {
    const float* x0    = (const float*)d_in[0];
    const int*   eidx  = (const int*)d_in[1];
    const float* eattr = (const float*)d_in[2];
    const int* e_src = eidx;            // edge_index[0]
    const int* e_dst = eidx + NE;       // edge_index[1]

    ConvP cv[4];
    for (int i = 0; i < 4; ++i) {
        const int b = 3 + i * 8;
        cv[i].Wsrc  = (const float*)d_in[b + 0];
        cv[i].Wdst  = (const float*)d_in[b + 1];
        cv[i].Wedge = (const float*)d_in[b + 2];
        cv[i].asrc  = (const float*)d_in[b + 3];
        cv[i].adst  = (const float*)d_in[b + 4];
        cv[i].aedge = (const float*)d_in[b + 5];
        cv[i].Wres  = (const float*)d_in[b + 6];
        cv[i].bias  = (const float*)d_in[b + 7];
    }
    const float* ln_dep_g  = (const float*)d_in[35];
    const float* ln_dep_b  = (const float*)d_in[36];
    const float* ln_dept_g = (const float*)d_in[37];
    const float* ln_dept_b = (const float*)d_in[38];

    // workspace carve-up (256B aligned)
    char* p = (char*)d_ws;
    auto carve = [&](size_t bytes) -> void* {
        void* r = (void*)p;
        p += (bytes + 255) & ~(size_t)255;
        return r;
    };
    _Float16* Wt     = (_Float16*)carve((size_t)256 * 128 * 2);
    float*    H      = (float*)carve((size_t)NN * 256 * 4);
    float*    aggr   = (float*)carve((size_t)NN * 128 * 4);
    float*    xdep   = (float*)carve((size_t)NN * 128 * 4);
    float*    xdpt   = (float*)carve((size_t)NN * 128 * 4);
    float*    a_src  = (float*)carve((size_t)NN * 4);
    float*    a_dst  = (float*)carve((size_t)NN * 4);
    float*    vdst   = (float*)carve(128 * 4);
    float*    kedge  = (float*)carve(256);
    float*    alpha  = (float*)carve((size_t)NE * 4);
    unsigned* maxbuf = (unsigned*)carve((size_t)NN * 4);
    float*    denom  = (float*)carve((size_t)NN * 4);

    auto run_conv = [&](const float* xin, const int* sp, const int* dp,
                        const ConvP& P, float* out, int stride_f4, int col_f4,
                        int do_ln, const float* g, const float* bb) {
        k_prep_wt<<<128, 256, 0, stream>>>(P.Wsrc, P.Wres, Wt);
        k_prep_attn<<<1, 128, 0, stream>>>(P.Wdst, P.adst, P.Wedge, P.aedge, vdst, kedge);
        k_gemm<<<NN / 16, 256, 0, stream>>>(xin, Wt, H);
        k_node_attn<<<NN / 8, 256, 0, stream>>>(xin, H, P.asrc, vdst, a_src, a_dst);
        k_init<<<NN * 128 / 256, 256, 0, stream>>>(aggr, denom, maxbuf);
        k_edge_alpha<<<NE / 256, 256, 0, stream>>>(sp, dp, eattr, a_src, a_dst,
                                                   kedge, alpha, maxbuf);
        k_edge_exp<<<NE / 256, 256, 0, stream>>>(dp, alpha, maxbuf, denom);
        k_edge_scatter<<<NE / 8, 256, 0, stream>>>(sp, dp, alpha, denom, H, aggr);
        k_node_out<<<NN / 8, 256, 0, stream>>>(aggr, H, P.bias, g, bb, do_ln,
                                               out, stride_f4, col_f4);
    };

    float* out = (float*)d_out;                    // [NN, 256]
    // layer 1: dep uses (src,dst); dept uses flipped edges
    run_conv(x0, e_src, e_dst, cv[0], xdep, 32, 0, 1, ln_dep_g, ln_dep_b);
    run_conv(x0, e_dst, e_src, cv[1], xdpt, 32, 0, 1, ln_dept_g, ln_dept_b);
    // layer 2: raw conv outputs concatenated into d_out
    run_conv(xdep, e_src, e_dst, cv[2], out, 64, 0,  0, ln_dep_g, ln_dep_b);
    run_conv(xdpt, e_dst, e_src, cv[3], out, 64, 32, 0, ln_dept_g, ln_dept_b);
}
